// MoeLayer_50955491999893
// MI455X (gfx1250) — matrix operand order repaired and checked by reading, not compiled, not run on hardware
//
#include <hip/hip_runtime.h>
#include <hip/hip_bf16.h>

// ---------------- problem constants ----------------
#define TOK_N   16384      // B*S
#define DIM_D   1024
#define DIM_H   2048
#define N_EXP   8

// ---------------- tiling ----------------
#define TILE_M  128        // tokens per expert block (8 x 16-row WMMA tiles held in VGPR accs)
#define THREADS 256        // 8 waves of 32
#define NWAVES  8
#define MT      8          // m-tiles per block (TILE_M/16)

#define DPAD    (DIM_D + 16)   // bf16 elems per x row in LDS (16B align kept, bank spread)
#define XS_BYTES   (TILE_M * DPAD * 2)
#define SMEM1_BYTES (XS_BYTES + TILE_M * 4)

// FFN2 chunking of h along K
#define KCH     64                       // K columns per staged chunk
#define NCHUNK  (DIM_H / KCH)            // 32
#define CHPAD   (KCH + 8)                // 72 bf16 per row (16B aligned, bank spread)
#define BUF_ELEMS (TILE_M * CHPAD)       // one buffer
#define SMEM2_BYTES (2 * BUF_ELEMS * 2 + TILE_M * 4 + TILE_M * 4)

typedef __attribute__((ext_vector_type(16))) __bf16 v16bf;
typedef __attribute__((ext_vector_type(8)))  __bf16 v8bf;
typedef __attribute__((ext_vector_type(8)))  float  v8f;

// ---------------- WMMA fragment helpers ----------------
// A (16x32 bf16) from row-major LDS tile: lane holds m=lane&15,
// K = kh*8+0..7 (regs 0-3) and K = 16+kh*8+0..7 (regs 4-7), kh=lane>>4.
__device__ __forceinline__ v16bf load_a_lds(const __bf16* tile, int stride, int lane) {
    const int m  = lane & 15;
    const int kh = lane >> 4;
    const __bf16* p = tile + m * stride + kh * 8;
    v8bf lo = *(const v8bf*)(p);
    v8bf hi = *(const v8bf*)(p + 16);
    v16bf a;
#pragma unroll
    for (int i = 0; i < 8; ++i) { a[i] = lo[i]; a[i + 8] = hi[i]; }
    return a;
}

// B (32x16 bf16): B[k][n] = W[n][k], W row-major [N x Kdim] fp32 in global.
// lane: n=lane&15, kg=lane>>4; K = kg*8+0..7 and 16+kg*8+0..7 -> 4x b128 + cvt_pk_bf16.
__device__ __forceinline__ v16bf load_b_global(const float* W, long long rowStride,
                                               int nbase, int k0, int lane) {
    const int n  = lane & 15;
    const int kg = lane >> 4;
    const float* p = W + (long long)(nbase + n) * rowStride + (k0 + kg * 8);
    const float4* p4 = (const float4*)p;
    float4 q0 = p4[0], q1 = p4[1], q2 = p4[4], q3 = p4[5];
    v16bf b;
    b[0]  = (__bf16)q0.x; b[1]  = (__bf16)q0.y; b[2]  = (__bf16)q0.z; b[3]  = (__bf16)q0.w;
    b[4]  = (__bf16)q1.x; b[5]  = (__bf16)q1.y; b[6]  = (__bf16)q1.z; b[7]  = (__bf16)q1.w;
    b[8]  = (__bf16)q2.x; b[9]  = (__bf16)q2.y; b[10] = (__bf16)q2.z; b[11] = (__bf16)q2.w;
    b[12] = (__bf16)q3.x; b[13] = (__bf16)q3.y; b[14] = (__bf16)q3.z; b[15] = (__bf16)q3.w;
    return b;
}

__device__ __forceinline__ v8f wmma_bf16(v16bf a, v16bf b, v8f c) {
    return __builtin_amdgcn_wmma_f32_16x16x32_bf16(false, a, false, b, (short)0, c, false, false);
}

// ---------------- kernel 1: zero output + counters ----------------
__global__ void moe_zero_kernel(float* __restrict__ out, long long n, int* __restrict__ counts) {
    long long i = (long long)blockIdx.x * blockDim.x + threadIdx.x;
    if (i < n) out[i] = 0.0f;
    if (i < N_EXP) counts[i] = 0;
}

// ---------------- kernel 2: gating (one wave per token) ----------------
__global__ void moe_gate_kernel(const float* __restrict__ x, const float* __restrict__ gate_w,
                                int* __restrict__ counts, int* __restrict__ selArr,
                                float* __restrict__ wArr) {
    __shared__ float gws[N_EXP * DIM_D];   // 32 KB
    const int tid = threadIdx.x;
    for (int i = tid; i < N_EXP * DIM_D; i += 256) gws[i] = gate_w[i];
    __syncthreads();

    const int wave = tid >> 5;
    const int lane = tid & 31;
    const int t = blockIdx.x * 8 + wave;

    float acc[N_EXP];
#pragma unroll
    for (int e = 0; e < N_EXP; ++e) acc[e] = 0.0f;

    const float* xr = x + (long long)t * DIM_D;
    for (int k = lane; k < DIM_D; k += 32) {
        float xv = xr[k];
#pragma unroll
        for (int e = 0; e < N_EXP; ++e) acc[e] += xv * gws[e * DIM_D + k];
    }
#pragma unroll
    for (int off = 16; off > 0; off >>= 1) {
#pragma unroll
        for (int e = 0; e < N_EXP; ++e) acc[e] += __shfl_xor(acc[e], off, 32);
    }

    if (lane == 0) {
        int e0 = 0; float l0 = acc[0];
#pragma unroll
        for (int e = 1; e < N_EXP; ++e) if (acc[e] > l0) { l0 = acc[e]; e0 = e; }
        int e1 = (e0 == 0) ? 1 : 0; float l1 = acc[e1];
#pragma unroll
        for (int e = 0; e < N_EXP; ++e)
            if (e != e0 && acc[e] > l1) { l1 = acc[e]; e1 = e; }
        float p1 = __expf(l1 - l0);
        float inv = 1.0f / (1.0f + p1);
        selArr[2 * t + 0] = e0;  wArr[2 * t + 0] = inv;
        selArr[2 * t + 1] = e1;  wArr[2 * t + 1] = p1 * inv;
        atomicAdd(&counts[e0], 1);
        atomicAdd(&counts[e1], 1);
    }
}

// ---------------- kernel 3: prefix offsets (E=8, trivial) ----------------
__global__ void moe_offsets_kernel(const int* __restrict__ counts,
                                   int* __restrict__ offs, int* __restrict__ cursor) {
    if (threadIdx.x == 0) {
        int acc = 0;
#pragma unroll
        for (int e = 0; e < N_EXP; ++e) { offs[e] = acc; acc += counts[e]; }
    }
    if (threadIdx.x < N_EXP) cursor[threadIdx.x] = 0;
}

// ---------------- kernel 4: scatter tokens into compact expert lists ----------------
__global__ void moe_scatter_kernel(const int* __restrict__ selArr, const float* __restrict__ wArr,
                                   const int* __restrict__ offs, int* __restrict__ cursor,
                                   int* __restrict__ tokList, float* __restrict__ wvList) {
    int t = blockIdx.x * blockDim.x + threadIdx.x;
    if (t >= TOK_N) return;
#pragma unroll
    for (int s = 0; s < 2; ++s) {
        int e = selArr[2 * t + s];
        int p = atomicAdd(&cursor[e], 1);
        int idx = offs[e] + p;
        tokList[idx] = t;
        wvList[idx]  = wArr[2 * t + s];
    }
}

// ---------------- kernel 5: FFN1  h = silu(x w1^T) * (x w3^T) -> ws (bf16) ----------------
__global__ __launch_bounds__(THREADS)
void moe_ffn1_kernel(const float* __restrict__ x,
                     const float* __restrict__ w1, const float* __restrict__ w3,
                     const int* __restrict__ counts, const int* __restrict__ offs,
                     const int* __restrict__ tokList, __bf16* __restrict__ hws) {
    extern __shared__ unsigned char smem[];
    __bf16* xs   = (__bf16*)smem;                  // [TILE_M][DPAD]
    int*    stok = (int*)(smem + XS_BYTES);        // [TILE_M]

    const int tilesPerE = TOK_N / TILE_M;          // 128
    const int e    = blockIdx.x / tilesPerE;       // blocks grouped by expert -> L2 locality
    const int tile = blockIdx.x % tilesPerE;
    const int cnt  = counts[e];
    const int row0 = tile * TILE_M;
    if (row0 >= cnt) return;
    const int vcnt  = min(TILE_M, cnt - row0);
    const int lbase = offs[e] + row0;              // global routed-row base for this tile

    const int tid  = threadIdx.x;
    const int wave = tid >> 5;
    const int lane = tid & 31;

    if (tid < TILE_M)
        stok[tid] = (tid < vcnt) ? tokList[lbase + tid] : 0;
    __syncthreads();

    // stage gathered x rows as bf16 into LDS (zeros for invalid rows -> h rows become 0)
    for (int idx = tid; idx < TILE_M * (DIM_D / 4); idx += THREADS) {
        int r  = idx / (DIM_D / 4);
        int c4 = idx % (DIM_D / 4);
        float4 v = make_float4(0.f, 0.f, 0.f, 0.f);
        if (r < vcnt)
            v = ((const float4*)(x + (long long)stok[r] * DIM_D))[c4];
        __bf16* dst = xs + r * DPAD + c4 * 4;
        dst[0] = (__bf16)v.x; dst[1] = (__bf16)v.y; dst[2] = (__bf16)v.z; dst[3] = (__bf16)v.w;
    }
    __syncthreads();

    const float* w1e = w1 + (long long)e * DIM_H * DIM_D;
    const float* w3e = w3 + (long long)e * DIM_H * DIM_D;
    const int NT = (DIM_H / 16) / NWAVES;          // 16 n-tiles per wave
    for (int j = 0; j < NT; ++j) {
        const int nbase = (wave * NT + j) * 16;
        v8f z = {};
        v8f acc1[MT], acc3[MT];
#pragma unroll
        for (int mt = 0; mt < MT; ++mt) { acc1[mt] = z; acc3[mt] = z; }

        for (int k0 = 0; k0 < DIM_D; k0 += 32) {
            v16bf b1 = load_b_global(w1e, DIM_D, nbase, k0, lane);
            v16bf b3 = load_b_global(w3e, DIM_D, nbase, k0, lane);
            if (k0 + 32 < DIM_D)
                __builtin_prefetch(w1e + (long long)(nbase + (lane & 15)) * DIM_D + k0 + 32, 0, 3);
#pragma unroll
            for (int mt = 0; mt < MT; ++mt) {       // 16 WMMAs per k-step, B reused 128 rows
                v16bf a = load_a_lds(xs + mt * 16 * DPAD + k0, DPAD, lane);
                acc1[mt] = wmma_bf16(a, b1, acc1[mt]);
                acc3[mt] = wmma_bf16(a, b3, acc3[mt]);
            }
        }
        // epilogue: SiLU*gate in fp32 -> bf16 h rows in workspace
        const int ncol = nbase + (lane & 15);
        const int kg   = lane >> 4;
#pragma unroll
        for (int mt = 0; mt < MT; ++mt) {
#pragma unroll
            for (int r = 0; r < 8; ++r) {
                float g  = acc1[mt][r];
                float hv = (g / (1.0f + __expf(-g))) * acc3[mt][r];
                long long grow = (long long)(lbase + mt * 16 + r + 8 * kg);
                hws[grow * DIM_H + ncol] = (__bf16)hv;
            }
        }
    }
}

// ---------------- kernel 6: FFN2  out += wv * (h w2^T), async-LDS h stream ----------------
__global__ __launch_bounds__(THREADS)
void moe_ffn2_kernel(const float* __restrict__ w2, const __bf16* __restrict__ hws,
                     const int* __restrict__ counts, const int* __restrict__ offs,
                     const int* __restrict__ tokList, const float* __restrict__ wvList,
                     float* __restrict__ out) {
    extern __shared__ unsigned char smem[];
    __bf16* hbuf = (__bf16*)smem;                              // [2][TILE_M][CHPAD]
    int*    stok = (int*)(smem + 2 * BUF_ELEMS * 2);           // [TILE_M]
    float*  swv  = (float*)(stok + TILE_M);                    // [TILE_M]

    const int tilesPerE = TOK_N / TILE_M;
    const int e    = blockIdx.x / tilesPerE;
    const int tile = blockIdx.x % tilesPerE;
    const int cnt  = counts[e];
    const int row0 = tile * TILE_M;
    if (row0 >= cnt) return;
    const int vcnt  = min(TILE_M, cnt - row0);
    const int lbase = offs[e] + row0;

    const int tid  = threadIdx.x;
    const int wave = tid >> 5;
    const int lane = tid & 31;

    if (tid < TILE_M) {
        if (tid < vcnt) { stok[tid] = tokList[lbase + tid]; swv[tid] = wvList[lbase + tid]; }
        else            { stok[tid] = 0;                    swv[tid] = 0.0f; }
    }

    const __bf16* hG = hws + (long long)lbase * DIM_H;     // this tile's 128 h rows
    const unsigned ldsBase = (unsigned)(uintptr_t)(void*)hbuf;
    const float* w2e = w2 + (long long)e * DIM_D * DIM_H;

    const int ROUNDS = DIM_D / (NWAVES * 16);              // 8 (block covers 128 cols/round)
    for (int rnd = 0; rnd < ROUNDS; ++rnd) {
        const int nbase = rnd * (NWAVES * 16) + wave * 16;
        v8f z = {};
        v8f acc[MT];
#pragma unroll
        for (int mt = 0; mt < MT; ++mt) acc[mt] = z;

        // prologue: async-stage chunk 0 (each thread DMAs 4x16B into LDS)
        {
#pragma unroll
            for (int q = 0; q < 4; ++q) {
                int seg = tid * 4 + q;                     // 0..1023
                int row = seg >> 3, s8 = seg & 7;
                const __bf16* gp = hG + (long long)row * DIM_H + s8 * 8;
                unsigned lo = ldsBase + (unsigned)(row * CHPAD + s8 * 8) * 2;
                asm volatile("global_load_async_to_lds_b128 %0, %1, off"
                             :: "v"(lo), "v"((unsigned long long)(uintptr_t)gp) : "memory");
            }
        }
        for (int c = 0; c < NCHUNK; ++c) {
            if (c + 1 < NCHUNK) {                          // stage next chunk, then wait current
                unsigned bsel = (unsigned)((c + 1) & 1) * (unsigned)(BUF_ELEMS * 2);
#pragma unroll
                for (int q = 0; q < 4; ++q) {
                    int seg = tid * 4 + q;
                    int row = seg >> 3, s8 = seg & 7;
                    const __bf16* gp = hG + (long long)row * DIM_H + (c + 1) * KCH + s8 * 8;
                    unsigned lo = ldsBase + bsel + (unsigned)(row * CHPAD + s8 * 8) * 2;
                    asm volatile("global_load_async_to_lds_b128 %0, %1, off"
                                 :: "v"(lo), "v"((unsigned long long)(uintptr_t)gp) : "memory");
                }
                asm volatile("s_wait_asynccnt 0x4" ::: "memory");
            } else {
                asm volatile("s_wait_asynccnt 0x0" ::: "memory");
            }
            __syncthreads();                               // chunk c visible to all waves

            const __bf16* cb = hbuf + (c & 1) * BUF_ELEMS;
#pragma unroll
            for (int ks = 0; ks < KCH / 32; ++ks) {        // 2 k-steps per chunk
                const int k0c = ks * 32;
                v16bf b = load_b_global(w2e, DIM_H, nbase, c * KCH + k0c, lane);
#pragma unroll
                for (int mt = 0; mt < MT; ++mt) {          // 8 WMMAs, B reused 128 rows
                    v16bf a = load_a_lds(cb + mt * 16 * CHPAD + k0c, CHPAD, lane);
                    acc[mt] = wmma_bf16(a, b, acc[mt]);
                }
            }
            __syncthreads();                               // buffer reusable for chunk c+2
        }
        // epilogue: weighted scatter-add (token touched by exactly its 2 experts)
        const int col = nbase + (lane & 15);
        const int kg  = lane >> 4;
#pragma unroll
        for (int mt = 0; mt < MT; ++mt) {
#pragma unroll
            for (int r = 0; r < 8; ++r) {
                int m = mt * 16 + r + 8 * kg;
                if (m < vcnt)
                    atomicAdd(&out[(long long)stok[m] * DIM_D + col], acc[mt][r] * swv[m]);
            }
        }
    }
}

// ---------------- host-side launch ----------------
extern "C" void kernel_launch(void* const* d_in, const int* in_sizes, int n_in,
                              void* d_out, int out_size, void* d_ws, size_t ws_size,
                              hipStream_t stream) {
    (void)in_sizes; (void)n_in; (void)out_size; (void)ws_size;
    const float* x  = (const float*)d_in[0];
    const float* gw = (const float*)d_in[1];
    const float* w1 = (const float*)d_in[2];
    const float* w3 = (const float*)d_in[3];
    const float* w2 = (const float*)d_in[4];
    float* out = (float*)d_out;

    unsigned char* ws = (unsigned char*)d_ws;
    int*    counts  = (int*)(ws + 0);
    int*    cursor  = (int*)(ws + 64);
    int*    offs    = (int*)(ws + 128);
    size_t  o = 1024;
    int*    selArr  = (int*)(ws + o);    o += (size_t)2 * TOK_N * 4;   // [2T]
    float*  wArr    = (float*)(ws + o);  o += (size_t)2 * TOK_N * 4;   // [2T]
    int*    tokList = (int*)(ws + o);    o += (size_t)2 * TOK_N * 4;   // [2T] compact
    float*  wvList  = (float*)(ws + o);  o += (size_t)2 * TOK_N * 4;   // [2T] compact
    o = (o + 1048575) & ~(size_t)1048575;
    __bf16* hws     = (__bf16*)(ws + o);                               // [2T][H] bf16, 128 MB

    const long long outN = (long long)TOK_N * DIM_D;
    moe_zero_kernel<<<(int)((outN + 255) / 256), 256, 0, stream>>>(out, outN, counts);
    moe_gate_kernel<<<TOK_N / 8, 256, 0, stream>>>(x, gw, counts, selArr, wArr);
    moe_offsets_kernel<<<1, 32, 0, stream>>>(counts, offs, cursor);
    moe_scatter_kernel<<<TOK_N / 256, 256, 0, stream>>>(selArr, wArr, offs, cursor,
                                                        tokList, wvList);
    moe_ffn1_kernel<<<N_EXP * (TOK_N / TILE_M), THREADS, SMEM1_BYTES, stream>>>(
        x, w1, w3, counts, offs, tokList, hws);
    moe_ffn2_kernel<<<N_EXP * (TOK_N / TILE_M), THREADS, SMEM2_BYTES, stream>>>(
        w2, hws, counts, offs, tokList, wvList, out);
}
